// TestHeteroGNN_1924145349232
// MI455X (gfx1250) — matrix operand order repaired and checked by reading, not compiled, not run on hardware
//
#include <hip/hip_runtime.h>
#include <hip/hip_bf16.h>

typedef __attribute__((ext_vector_type(8)))  float          v8f;
typedef __attribute__((ext_vector_type(16))) __bf16         v16bf;

// round-to-nearest-even float -> bf16 bits (used only in one-time weight pack)
__device__ __forceinline__ unsigned short f2bf(float f) {
    unsigned u = __builtin_bit_cast(unsigned, f);
    u += 0x7FFFu + ((u >> 16) & 1u);
    return (unsigned short)(u >> 16);
}

// ---------------- kernel 0: zero accumulators ----------------
__global__ void zero_ws_kernel(float* __restrict__ p, long long n) {
    long long i = (long long)blockIdx.x * blockDim.x + threadIdx.x;
    if (i < n) p[i] = 0.0f;
}

// ---------------- kernel 1: pack weights into WMMA B-fragment layout (bf16) ----
// wpack element g: j=g&15, lane=(g>>4)&31, ntile=(g>>9)&7, kc=(g>>12)&1, mat=g>>13
// B order used here (contiguous k; A matches): lane l -> N = l%16, elem j -> K = (l/16)*16 + j
__global__ void pack_w_kernel(const float* __restrict__ Wl,
                              const float* __restrict__ Wr,
                              unsigned short* __restrict__ wp) {
    int g = blockIdx.x * blockDim.x + threadIdx.x;
    if (g >= 16384) return;
    int j    = g & 15;
    int lane = (g >> 4) & 31;
    int nt   = (g >> 9) & 7;
    int kc   = (g >> 12) & 1;
    int mat  = g >> 13;
    int h    = lane >> 4;
    int K    = kc * 32 + h * 16 + j;
    int ncol = nt * 16 + (lane & 15);
    const float* W = mat ? Wr : Wl;
    wp[g] = f2bf(W[K * 128 + ncol]);
}

// ---------------- kernel 2: edge gather + atomic scatter-mean accumulate ------
// 16 threads per edge, float4 (16B) per thread = 64 floats per edge
__global__ void edge_scatter_kernel(const float* __restrict__ xprot,
                                    const int* __restrict__ src,
                                    const int* __restrict__ dst,
                                    float* __restrict__ agg,
                                    float* __restrict__ cnt,
                                    int E) {
    long long gid = (long long)blockIdx.x * blockDim.x + threadIdx.x;
    if (gid >= (long long)E * 16) return;
    int e = (int)(gid >> 4);
    int c = (int)(gid & 15);
    int s = src[e];
    int d = dst[e];
    float4 v = ((const float4*)xprot)[(size_t)s * 16 + c];
    float* a = agg + (size_t)d * 64 + c * 4;
    unsafeAtomicAdd(a + 0, v.x);
    unsafeAtomicAdd(a + 1, v.y);
    unsafeAtomicAdd(a + 2, v.z);
    unsafeAtomicAdd(a + 3, v.w);
    if (c == 0) unsafeAtomicAdd(&cnt[d], 1.0f);
}

// ---------------- kernel 3: fused normalize + dual GEMM (WMMA bf16) + relu + reduce
// block = 256 threads = 8 waves; EACH WAVE owns one 16-row M-tile and sweeps all
// 8 N-tiles (H=128) with 8 persistent f32 accumulators, so the A fragments are
// converted to bf16 exactly once per row tile and reused across 8 WMMAs.
// A order used here (contiguous k; matches wpack B), 16x32 MxK: lane l -> M = l%16,
//   elem j -> K = (l/16)*16 + j
__global__ void gemm_reduce_kernel(const float* __restrict__ agg,
                                   const float* __restrict__ cnt,
                                   const float* __restrict__ xlig,
                                   const unsigned short* __restrict__ wpack,
                                   const float* __restrict__ bias,
                                   float* __restrict__ colsum,
                                   int n_lig) {
    int lane = threadIdx.x & 31;
    int wv   = threadIdx.x >> 5;
    int rowtile = blockIdx.x * 8 + wv;
    int rowbase = rowtile * 16;
    if (rowbase >= n_lig) return;           // uniform across the wave

    int r = lane & 15;
    int h = lane >> 4;
    int row = rowbase + r;
    bool rvalid = row < n_lig;
    int rowc = rvalid ? row : (n_lig - 1);  // clamped; contribution masked below

    float cr = cnt[rowc];
    float scale = 1.0f / fmaxf(cr, 1.0f);

    const float4* aggp = (const float4*)(agg  + (size_t)rowc * 64);
    const float4* xp   = (const float4*)(xlig + (size_t)rowc * 64);

    v8f acc[8];
#pragma unroll
    for (int nt = 0; nt < 8; ++nt) acc[nt] = (v8f){0.f,0.f,0.f,0.f,0.f,0.f,0.f,0.f};

#pragma unroll
    for (int mat = 0; mat < 2; ++mat) {
        const float4* srcp = (mat == 0) ? aggp : xp;
        float s = ((mat == 0) ? scale : 1.0f) * (rvalid ? 1.0f : 0.0f);
#pragma unroll
        for (int kc = 0; kc < 2; ++kc) {
            int base4 = kc * 8 + h * 4;      // float4 index of K = kc*32 + h*16
            float4 a0 = srcp[base4];
            float4 a1 = srcp[base4 + 1];
            float4 a2 = srcp[base4 + 2];     // K = kc*32 + h*16 + 8
            float4 a3 = srcp[base4 + 3];
            v16bf av;
            av[0]  = (__bf16)(a0.x * s); av[1]  = (__bf16)(a0.y * s);
            av[2]  = (__bf16)(a0.z * s); av[3]  = (__bf16)(a0.w * s);
            av[4]  = (__bf16)(a1.x * s); av[5]  = (__bf16)(a1.y * s);
            av[6]  = (__bf16)(a1.z * s); av[7]  = (__bf16)(a1.w * s);
            av[8]  = (__bf16)(a2.x * s); av[9]  = (__bf16)(a2.y * s);
            av[10] = (__bf16)(a2.z * s); av[11] = (__bf16)(a2.w * s);
            av[12] = (__bf16)(a3.x * s); av[13] = (__bf16)(a3.y * s);
            av[14] = (__bf16)(a3.z * s); av[15] = (__bf16)(a3.w * s);

            const unsigned short* wb =
                wpack + (size_t)((mat * 2 + kc) * 8) * 512 + (size_t)lane * 16;
#pragma unroll
            for (int nt = 0; nt < 8; ++nt) {
                v16bf bv = *(const v16bf*)(wb + (size_t)nt * 512);
                acc[nt] = __builtin_amdgcn_wmma_f32_16x16x32_bf16(
                    false, av, false, bv, (short)0, acc[nt], false, false);
            }
        }
    }

    // C layout: elem i -> M = h*8 + i, N = lane%16. Bias + relu + row-sum,
    // combine the two M-halves (lanes l and l+16 share N) before the atomic.
#pragma unroll
    for (int nt = 0; nt < 8; ++nt) {
        int ncol = nt * 16 + r;
        float b = bias[ncol];
        float ssum = 0.0f;
#pragma unroll
        for (int i = 0; i < 8; ++i) {
            int m = rowbase + h * 8 + i;
            float v = fmaxf(acc[nt][i] + b, 0.0f);
            ssum += (m < n_lig) ? v : 0.0f;
        }
        ssum += __shfl_down(ssum, 16, 32);
        if (h == 0) unsafeAtomicAdd(&colsum[ncol], ssum);
    }
}

// ---------------- kernel 4: finalize ----------------
__global__ void finalize_kernel(const float* __restrict__ colsum,
                                const float* __restrict__ W_lin,
                                const float* __restrict__ b_lin,
                                float* __restrict__ out,
                                int n_lig) {
    __shared__ float sh[128];
    int t = threadIdx.x;
    sh[t] = colsum[t] * W_lin[t] * (1.0f / (float)n_lig);
    __syncthreads();
    for (int s = 64; s > 0; s >>= 1) {
        if (t < s) sh[t] += sh[t + s];
        __syncthreads();
    }
    if (t == 0) out[0] = sh[0] + b_lin[0];
}

extern "C" void kernel_launch(void* const* d_in, const int* in_sizes, int n_in,
                              void* d_out, int out_size, void* d_ws, size_t ws_size,
                              hipStream_t stream) {
    const float* x_lig  = (const float*)d_in[0];
    const float* x_prot = (const float*)d_in[1];
    // d_in[2..4]: W_l_lp, b_lp, W_r_lp -> dead branch, unused
    const float* W_l_pl = (const float*)d_in[5];
    const float* b_pl   = (const float*)d_in[6];
    const float* W_r_pl = (const float*)d_in[7];
    const float* W_lin  = (const float*)d_in[8];
    const float* b_lin  = (const float*)d_in[9];
    // d_in[10..11]: src_lp, dst_lp -> dead branch, unused
    const int* src_pl = (const int*)d_in[12];
    const int* dst_pl = (const int*)d_in[13];

    const int D = 64, H = 128;
    const int n_lig = in_sizes[0] / D;
    const int E     = in_sizes[12];

    // workspace layout: [agg n_lig*64 f32][cnt n_lig f32][colsum 128 f32][wpack 16384 bf16]
    float* agg    = (float*)d_ws;
    float* cnt    = agg + (size_t)n_lig * D;
    float* colsum = cnt + n_lig;
    size_t wp_off = (((size_t)n_lig * D + n_lig + H) * sizeof(float) + 31) & ~(size_t)31;
    unsigned short* wpack = (unsigned short*)((char*)d_ws + wp_off);

    // 0) zero accumulators (every call: graph-replay determinism)
    long long nz = (long long)n_lig * D + n_lig + H;
    zero_ws_kernel<<<(int)((nz + 255) / 256), 256, 0, stream>>>((float*)d_ws, nz);

    // 1) pack weights to WMMA B layout
    pack_w_kernel<<<(16384 + 255) / 256, 256, 0, stream>>>(W_l_pl, W_r_pl, wpack);

    // 2) edge gather + atomic scatter
    long long nthr = (long long)E * 16;
    edge_scatter_kernel<<<(int)((nthr + 255) / 256), 256, 0, stream>>>(
        x_prot, src_pl, dst_pl, agg, cnt, E);

    // 3) fused normalize + dual-GEMM (8 waves x 16-row tiles, 32 WMMA/wave) + reduce
    int rtiles = (n_lig + 15) / 16;
    int blocks = (rtiles + 7) / 8;
    gemm_reduce_kernel<<<blocks, 256, 0, stream>>>(
        agg, cnt, x_lig, wpack, b_pl, colsum, n_lig);

    // 4) finalize scalar output
    finalize_kernel<<<1, 128, 0, stream>>>(colsum, W_lin, b_lin, (float*)d_out, n_lig);
}
